// TestNet55_descv2_54142357733428
// MI455X (gfx1250) — matrix operand order repaired and checked by reading, not compiled, not run on hardware
//
#include <hip/hip_runtime.h>

#define NN 150000
#define EE 900000
#define BB 64
#define EPSV 1e-5f

typedef _Float16 h16_t;
typedef __attribute__((ext_vector_type(16))) _Float16 v16h;
typedef __attribute__((ext_vector_type(8)))  _Float16 v8h;
typedef __attribute__((ext_vector_type(8)))  float    v8f;

#define SBH 264   // padded LDS stride in halves (528B = 33*16 -> aligned, bank-skewed)

// ---------------- utility kernels ----------------

__global__ void k_fill_f32(float* __restrict__ p, long long n, float v) {
    long long i = (long long)blockIdx.x * blockDim.x + threadIdx.x;
    long long st = (long long)gridDim.x * blockDim.x;
    for (; i < n; i += st) p[i] = v;
}

__global__ void k_deg(const int* __restrict__ dst, float* __restrict__ deg, int E) {
    int i = blockIdx.x * blockDim.x + threadIdx.x;
    if (i < E) atomicAdd(&deg[dst[i]], 1.0f);
}

__global__ void k_dinv(float* __restrict__ d, int n) {
    int i = blockIdx.x * blockDim.x + threadIdx.x;
    if (i < n) d[i] = rsqrtf(d[i] + 1.0f);
}

__global__ void k_ce(const float* __restrict__ dinv, const int* __restrict__ src,
                     const int* __restrict__ dst, float* __restrict__ ce, int E) {
    int i = blockIdx.x * blockDim.x + threadIdx.x;
    if (i < E) ce[i] = dinv[src[i]] * dinv[dst[i]];
}

// pos (N,3) f32 -> x16 (N,32) f16 zero padded
__global__ void k_pos(const float* __restrict__ pos, h16_t* __restrict__ x, int n) {
    int i = blockIdx.x, c = threadIdx.x;
    if (i < n) x[(size_t)i * 32 + c] = (c < 3) ? (h16_t)pos[(size_t)i * 3 + c] : (h16_t)0.f;
}

// W (din,dout) f32 -> Wh (Kp,dout) f16, rows >= din zeroed.  grid = (1, Kp)
__global__ void k_padw(const float* __restrict__ W, h16_t* __restrict__ Wh,
                       int din, int dout) {
    int k = blockIdx.y;
    for (int c = threadIdx.x; c < dout; c += blockDim.x)
        Wh[(size_t)k * dout + c] = (k < din) ? (h16_t)W[(size_t)k * dout + c] : (h16_t)0.f;
}

// ---------------- WMMA GEMM ----------------
// O = A(MxKp,f16) @ Bw(Kpxdout,f16) [+ bias] [relu]
// 256 threads = 8 waves; each wave owns TWO 16-row tiles (block = 256 rows),
// sharing one B fragment per K-step across both WMMAs.
// B tile staged in LDS transposed ([n][k], padded stride) so each lane's
// fragment is two 16-byte runs, 16 halves apart -> 2x ds_load_b128.
// A rows are CLAMPED (not predicated) so the K-loop is branch-free; rows >= M
// produce garbage accumulators that are simply never stored.
__global__ void __launch_bounds__(256) k_gemm(
    const h16_t* __restrict__ A, int Kp,
    const h16_t* __restrict__ Bw, int dout, int DP,
    h16_t* __restrict__ O16, float* __restrict__ O32,
    const float* __restrict__ bias, int relu, int M)
{
    __shared__ __align__(32) h16_t sB[16 * SBH];
    const int tid  = threadIdx.x;
    const int lane = tid & 31, wave = tid >> 5;
    const int nn   = lane & 15, hi = lane >> 4;
    const int rowBase = blockIdx.x * 256 + wave * 32;   // two 16-row tiles
    const int nk = Kp >> 5;                             // K tiles of 32
    const int ntiles = DP >> 4;                         // N tiles of 16

    // clamped A row pointers (always in-bounds)
    int r0 = rowBase + nn;      if (r0 >= M) r0 = M - 1;
    int r1 = rowBase + 16 + nn; if (r1 >= M) r1 = M - 1;
    const h16_t* ap0 = A + (size_t)r0 * Kp + hi * 8;
    const h16_t* ap1 = A + (size_t)r1 * Kp + hi * 8;

    for (int nt = 0; nt < ntiles; ++nt) {
        // stage B(:, nt*16 .. nt*16+15) into LDS transposed: sB[n][k]
        for (int idx = tid; idx < Kp * 16; idx += 256) {
            int k = idx >> 4;
            int n = idx & 15;
            int c = n + nt * 16;
            sB[n * SBH + k] = (c < dout) ? Bw[(size_t)k * dout + c] : (h16_t)0.f;
        }
        __syncthreads();

        v8f acc0 = {}, acc1 = {};
        for (int kt = 0; kt < nk; ++kt) {
            // B fragment: 32x16, lane col = nn, K = kt*32 + hi*8 + e (e<8), kt*32 + 16 + hi*8 + (e-8) (e>=8)
            const h16_t* bp = sB + nn * SBH + kt * 32 + hi * 8;
            v8h b0 = *(const v8h*)(bp);
            v8h b1 = *(const v8h*)(bp + 16);
            v16h b = __builtin_shufflevector(b0, b1,
                0, 1, 2, 3, 4, 5, 6, 7, 8, 9, 10, 11, 12, 13, 14, 15);

            // A fragments: 16x32, lane row = nn, K halves chosen by hi
            const h16_t* a0p = ap0 + kt * 32;
            v8h a00 = *(const v8h*)(a0p);
            v8h a01 = *(const v8h*)(a0p + 16);
            v16h a0 = __builtin_shufflevector(a00, a01,
                0, 1, 2, 3, 4, 5, 6, 7, 8, 9, 10, 11, 12, 13, 14, 15);

            const h16_t* a1p = ap1 + kt * 32;
            v8h a10 = *(const v8h*)(a1p);
            v8h a11 = *(const v8h*)(a1p + 16);
            v16h a1 = __builtin_shufflevector(a10, a11,
                0, 1, 2, 3, 4, 5, 6, 7, 8, 9, 10, 11, 12, 13, 14, 15);

            acc0 = __builtin_amdgcn_wmma_f32_16x16x32_f16(
                false, a0, false, b, (short)0, acc0, false, false);
            acc1 = __builtin_amdgcn_wmma_f32_16x16x32_f16(
                false, a1, false, b, (short)0, acc1, false, false);
        }

        // store D: row = tileBase + hi*8 + r, col = nt*16 + nn
        int col = nt * 16 + nn;
        float bv = (bias != nullptr && col < dout) ? bias[col] : 0.f;
#pragma unroll
        for (int r = 0; r < 8; ++r) {
            int row = rowBase + hi * 8 + r;
            if (row < M) {
                float v = acc0[r];
                if (col < dout) { v += bv; if (relu) v = fmaxf(v, 0.f); }
                else v = 0.f;
                if (O16) O16[(size_t)row * DP + col] = (h16_t)v;
                if (O32) O32[(size_t)row * DP + col] = v;
            }
        }
#pragma unroll
        for (int r = 0; r < 8; ++r) {
            int row = rowBase + 16 + hi * 8 + r;
            if (row < M) {
                float v = acc1[r];
                if (col < dout) { v += bv; if (relu) v = fmaxf(v, 0.f); }
                else v = 0.f;
                if (O16) O16[(size_t)row * DP + col] = (h16_t)v;
                if (O32) O32[(size_t)row * DP + col] = v;
            }
        }
        __syncthreads();
    }
}

// ---------------- graph aggregation ----------------
// one block per edge, one thread per channel: agg[dst] += h[src]*ce
__global__ void k_scatter(const h16_t* __restrict__ h, const float* __restrict__ ce,
                          const int* __restrict__ src, const int* __restrict__ dst,
                          float* __restrict__ agg, int dout, int DP) {
    int e = blockIdx.x;
    int c = threadIdx.x;
    int s = src[e], d = dst[e];
    float w = ce[e];
    __builtin_prefetch(h + (size_t)s * DP, 0, 1);
    if (c < dout) {
        float v = (float)h[(size_t)s * DP + c] * w;
        atomicAdd(&agg[(size_t)d * DP + c], v);
    }
}

// z = agg + h*cs + b (in place), accumulate per-channel sum/sumsq for BN
// blockDim = DP (thread = channel), each block covers 64 rows
__global__ void k_self(const h16_t* __restrict__ h, const float* __restrict__ dinv,
                       const float* __restrict__ bias, float* __restrict__ z,
                       float* __restrict__ sum, float* __restrict__ sumsq,
                       int n, int dout, int DP) {
    int c = threadIdx.x;
    if (c >= dout) return;
    int r0 = blockIdx.x * 64;
    float bc = bias[c];
    float s = 0.f, s2 = 0.f;
    for (int r = 0; r < 64; ++r) {
        int i = r0 + r;
        if (i >= n) break;
        float di = dinv[i];
        size_t o = (size_t)i * DP + c;
        float v = z[o] + (float)h[o] * (di * di) + bc;
        z[o] = v;
        s += v; s2 += v * v;
    }
    atomicAdd(&sum[c], s);
    atomicAdd(&sumsq[c], s2);
}

__global__ void k_bnfin(const float* __restrict__ sum, const float* __restrict__ sumsq,
                        const float* __restrict__ g, const float* __restrict__ be,
                        float* __restrict__ scA, float* __restrict__ shB,
                        int dout, float invN) {
    int c = blockIdx.x * blockDim.x + threadIdx.x;
    if (c < dout) {
        float m  = sum[c] * invN;
        float v  = sumsq[c] * invN - m * m;
        float rs = rsqrtf(v + EPSV);
        float sc = g[c] * rs;
        scA[c] = sc;
        shB[c] = be[c] - m * sc;
    }
}

// relu(z*scA+shB) -> f16 (padding channels zeroed). grid = N rows, block = DP
__global__ void k_apply(const float* __restrict__ z, const float* __restrict__ scA,
                        const float* __restrict__ shB, h16_t* __restrict__ x,
                        int dout, int DP) {
    int i = blockIdx.x, c = threadIdx.x;
    float v = 0.f;
    if (c < dout) v = fmaxf(z[(size_t)i * DP + c] * scA[c] + shB[c], 0.f);
    x[(size_t)i * DP + c] = (h16_t)v;
}

// segment max over sorted equal partition batch = (i*B)//N.  grid=B, block=128
__global__ void k_segmax(const float* __restrict__ X, h16_t* __restrict__ P, int n) {
    int b = blockIdx.x, c = threadIdx.x;
    int start = (b * n + BB - 1) / BB;
    int end   = ((b + 1) * n + BB - 1) / BB;
    float m = -3.4e38f;
    for (int i = start; i < end; ++i)
        m = fmaxf(m, X[(size_t)i * 128 + c]);
    P[b * 128 + c] = (h16_t)m;
}

// row-wise L2 normalize (B x 128). grid=B, block=128
__global__ void k_l2(const float* __restrict__ X, float* __restrict__ O) {
    __shared__ float red[128];
    int b = blockIdx.x, c = threadIdx.x;
    float v = X[b * 128 + c];
    red[c] = v * v;
    __syncthreads();
    for (int s = 64; s > 0; s >>= 1) {
        if (c < s) red[c] += red[c + s];
        __syncthreads();
    }
    float nrm = sqrtf(red[0]);
    O[b * 128 + c] = v / fmaxf(nrm, 1e-12f);
}

// ---------------- host ----------------

extern "C" void kernel_launch(void* const* d_in, const int* in_sizes, int n_in,
                              void* d_out, int out_size, void* d_ws, size_t ws_size,
                              hipStream_t stream) {
    (void)in_sizes; (void)n_in; (void)out_size; (void)ws_size;
    const int N = NN, E = EE;
    const float* pos = (const float*)d_in[0];
    const int* src = (const int*)d_in[1];
    const int* dst = src + E;
    // d_in[2] = batch (sorted equal partition; bounds computed analytically)
    const float *W[5], *bias[5], *gam[5], *bet[5];
    int p = 3;
    for (int l = 0; l < 5; ++l) {
        W[l]    = (const float*)d_in[p++];
        bias[l] = (const float*)d_in[p++];
        gam[l]  = (const float*)d_in[p++];
        bet[l]  = (const float*)d_in[p++];
    }
    const float *fcW[4], *fcb[4];
    for (int l = 0; l < 4; ++l) {
        fcW[l] = (const float*)d_in[p++];
        fcb[l] = (const float*)d_in[p++];
    }

    // workspace layout
    char* wsb = (char*)d_ws;
    size_t off = 0;
    auto walloc = [&](size_t bytes) -> void* {
        void* r = wsb + off;
        off += (bytes + 255) & ~(size_t)255;
        return r;
    };
    h16_t* x16 = (h16_t*)walloc((size_t)N * 256 * 2);     // GEMM input (padded K)
    h16_t* hbf = (h16_t*)walloc((size_t)N * 256 * 2);     // GEMM output h
    float* agg = (float*)walloc((size_t)N * 256 * 4);     // aggregation / z / fc0 out
    float* ce  = (float*)walloc((size_t)E * 4);
    float* dinv= (float*)walloc((size_t)N * 4);
    float* sums= (float*)walloc(1024 * 4);                // sum|sumsq|scA|shB
    float* sumsq = sums + 256;
    float* scA   = sums + 512;
    float* shB   = sums + 768;
    h16_t* wh  = (h16_t*)walloc(256 * 1024);              // f16 weights (all layers)
    h16_t* pooled = (h16_t*)walloc(BB * 128 * 2);
    h16_t* fcA = (h16_t*)walloc(BB * 128 * 2);
    h16_t* fcB = (h16_t*)walloc(BB * 128 * 2);
    float* fco = (float*)walloc(BB * 128 * 4);

    // GCN normalization coefficients
    k_fill_f32<<<256, 256, 0, stream>>>(dinv, N, 0.f);
    k_deg<<<(E + 255) / 256, 256, 0, stream>>>(dst, dinv, E);
    k_dinv<<<(N + 255) / 256, 256, 0, stream>>>(dinv, N);
    k_ce<<<(E + 255) / 256, 256, 0, stream>>>(dinv, src, dst, ce, E);
    k_pos<<<N, 32, 0, stream>>>(pos, x16, N);

    const int din[5] = {3, 16, 32, 64, 94};
    const int dou[5] = {16, 32, 64, 94, 256};
    size_t woff = 0;
    for (int l = 0; l < 5; ++l) {
        int Kp = (din[l] + 31) & ~31;
        int DP = (dou[l] + 31) & ~31;
        h16_t* Whl = wh + woff; woff += (size_t)Kp * dou[l];
        k_padw<<<dim3(1, Kp), 256, 0, stream>>>(W[l], Whl, din[l], dou[l]);
        k_gemm<<<(N + 255) / 256, 256, 0, stream>>>(x16, Kp, Whl, dou[l], DP,
                                                    hbf, nullptr, nullptr, 0, N);
        k_fill_f32<<<2048, 256, 0, stream>>>(agg, (long long)N * DP, 0.f);
        k_scatter<<<E, DP, 0, stream>>>(hbf, ce, src, dst, agg, dou[l], DP);
        k_fill_f32<<<1, 512, 0, stream>>>(sums, 512, 0.f);
        k_self<<<(N + 63) / 64, DP, 0, stream>>>(hbf, dinv, bias[l], agg,
                                                 sums, sumsq, N, dou[l], DP);
        k_bnfin<<<1, 256, 0, stream>>>(sums, sumsq, gam[l], bet[l], scA, shB,
                                       dou[l], 1.f / (float)N);
        k_apply<<<N, DP, 0, stream>>>(agg, scA, shB, x16, dou[l], DP);
    }

    // fc0: (N,256)@(256,128)+b, relu -> f32 (reuse agg, stride 128)
    {
        h16_t* Wf = wh + woff; woff += (size_t)256 * 128;
        k_padw<<<dim3(1, 256), 256, 0, stream>>>(fcW[0], Wf, 256, 128);
        k_gemm<<<(N + 255) / 256, 256, 0, stream>>>(x16, 256, Wf, 128, 128,
                                                    nullptr, agg, fcb[0], 1, N);
    }
    k_segmax<<<BB, 128, 0, stream>>>(agg, pooled, N);

    // fc1, fc2 (relu, f16 out), fc3 (no relu, f32 out)
    h16_t* Wf1 = wh + woff; woff += (size_t)128 * 128;
    k_padw<<<dim3(1, 128), 256, 0, stream>>>(fcW[1], Wf1, 128, 128);
    k_gemm<<<1, 256, 0, stream>>>(pooled, 128, Wf1, 128, 128, fcA, nullptr, fcb[1], 1, BB);

    h16_t* Wf2 = wh + woff; woff += (size_t)128 * 128;
    k_padw<<<dim3(1, 128), 256, 0, stream>>>(fcW[2], Wf2, 128, 128);
    k_gemm<<<1, 256, 0, stream>>>(fcA, 128, Wf2, 128, 128, fcB, nullptr, fcb[2], 1, BB);

    h16_t* Wf3 = wh + woff; woff += (size_t)128 * 128;
    k_padw<<<dim3(1, 128), 256, 0, stream>>>(fcW[3], Wf3, 128, 128);
    k_gemm<<<1, 256, 0, stream>>>(fcB, 128, Wf3, 128, 128, nullptr, fco, fcb[3], 0, BB);

    k_l2<<<BB, 128, 0, stream>>>(fco, (float*)d_out);
}
